// KANAutoencoder_5325759447261
// MI455X (gfx1250) — matrix operand order repaired and checked by reading, not compiled, not run on hardware
//
#include <hip/hip_runtime.h>
#include <hip/hip_bf16.h>

typedef __bf16 bf16_t;
typedef __attribute__((ext_vector_type(16))) __bf16 v16bf;
typedef __attribute__((ext_vector_type(8)))  float  v8f;

// exp(-0.5*((x-c)/0.4)^2) = exp2(KEXP2 * (x-c)^2),  KEXP2 = -0.5*6.25*log2(e)
#define KEXP2 (-4.5084225f)

// Deterministic hash -> standard normal (JAX threefry not reproducible; sn=0.01)
__device__ __forceinline__ float gauss_noise(unsigned seed, unsigned idx) {
  unsigned s = idx ^ (seed * 0x9E3779B9u);
  unsigned a = s * 747796405u + 2891336453u;
  a = ((a >> ((a >> 28u) + 4u)) ^ a) * 277803737u;
  a = (a >> 22u) ^ a;
  unsigned b = (s + 0x68bc21ebu) * 747796405u + 2891336453u;
  b = ((b >> ((b >> 28u) + 4u)) ^ b) * 277803737u;
  b = (b >> 22u) ^ b;
  float u1 = (a >> 8) * 5.9604645e-08f + 5.9604645e-08f;  // (0,1]
  float u2 = (b >> 8) * 5.9604645e-08f;
  float r = __fsqrt_rn(-2.0f * __logf(u1));
  return r * __cosf(6.2831853f * u2);
}

// One KAN layer: OUT[b,o] = silu(x@bw)*sb + (gauss_bases(x)@sw)*ss + noise*sn
// WG = 256 threads = 8 waves; WG tile 64(M) x 128(N); wave tile 32x32.
// K-loop: 16 input features / iter  (spline K-chunk = 128, base K-chunk = 16).
__global__ __launch_bounds__(256) void kan_layer_kernel(
    const float* __restrict__ X,   // [B, I]
    const float* __restrict__ SW,  // [I, O, 8]
    const float* __restrict__ BW,  // [I, O]
    const float* __restrict__ SB,  // [1, O]
    const float* __restrict__ SS,  // [1, O]
    const float* __restrict__ SN,  // scalar
    float* __restrict__ OUT,       // [B, O]
    int I, int O, unsigned seed) {
  __shared__ float  lds_x[64][16];      // x tile (f32), rows 64B (b128-aligned)
  __shared__ bf16_t lds_bw[128][16];    // bw tile, [n][k]
  __shared__ bf16_t lds_sw[128][136];   // sw tile, [n][k=i*8+g], padded rows
  __shared__ bf16_t lds_bas[64][136];   // gaussian-basis tile, [m][k], padded rows

  const int tid   = threadIdx.x;
  const int lane  = tid & 31;
  const int wave  = tid >> 5;
  const int wm    = (wave & 1) * 32;   // wave M offset in WG tile
  const int wn    = (wave >> 1) * 32;  // wave N offset in WG tile
  const int lhalf = (lane >= 16) ? 1 : 0;
  const int l16   = lane & 15;
  const int blockM = blockIdx.x * 64;
  const int blockN = blockIdx.y * 128;

  v8f acc_s[2][2], acc_b[2][2];
#pragma unroll
  for (int mt = 0; mt < 2; ++mt)
#pragma unroll
    for (int nt = 0; nt < 2; ++nt)
#pragma unroll
      for (int k = 0; k < 8; ++k) { acc_s[mt][nt][k] = 0.0f; acc_b[mt][nt][k] = 0.0f; }

  // per-thread fill roles (constant across K-loop)
  const int xrow = tid >> 2;          // 0..63
  const int xq   = tid & 3;           // feature quadrant (4 feats each)

  for (int i0 = 0; i0 < I; i0 += 16) {
    __syncthreads();
    { // x tile via CDNA5 async global->LDS DMA (f32, no conversion needed)
      const float* gp = &X[(size_t)(blockM + xrow) * I + i0 + xq * 4];
      unsigned lp = (unsigned)(unsigned long long)&lds_x[xrow][xq * 4];
      asm volatile("global_load_async_to_lds_b128 %0, %1, off"
                   :: "v"(lp), "v"(gp) : "memory");
    }
    { // bw tile: 16 x 128 f32 -> bf16, stored transposed [n][i]
      int i  = tid >> 4;
      int n8 = (tid & 15) * 8;
      const float* src = &BW[(size_t)(i0 + i) * O + blockN + n8];
      float4 a = *(const float4*)(src);
      float4 b = *(const float4*)(src + 4);
      lds_bw[n8 + 0][i] = (bf16_t)a.x; lds_bw[n8 + 1][i] = (bf16_t)a.y;
      lds_bw[n8 + 2][i] = (bf16_t)a.z; lds_bw[n8 + 3][i] = (bf16_t)a.w;
      lds_bw[n8 + 4][i] = (bf16_t)b.x; lds_bw[n8 + 5][i] = (bf16_t)b.y;
      lds_bw[n8 + 6][i] = (bf16_t)b.z; lds_bw[n8 + 7][i] = (bf16_t)b.w;
    }
#pragma unroll
    for (int j = 0; j < 8; ++j) { // sw tile: 16 i x 128 n x 8 g
      int p = tid + 256 * j;      // (i,n) pair index, 2048 total
      int i = p >> 7;
      int n = p & 127;
      const float* src = &SW[(((size_t)(i0 + i) * O) + blockN + n) * 8];
      float4 a = *(const float4*)(src);
      float4 b = *(const float4*)(src + 4);
      union { bf16_t h[8]; uint4 q; } w;
      w.h[0] = (bf16_t)a.x; w.h[1] = (bf16_t)a.y; w.h[2] = (bf16_t)a.z; w.h[3] = (bf16_t)a.w;
      w.h[4] = (bf16_t)b.x; w.h[5] = (bf16_t)b.y; w.h[6] = (bf16_t)b.z; w.h[7] = (bf16_t)b.w;
      *(uint4*)&lds_sw[n][i * 8] = w.q;
    }
    asm volatile("s_wait_asynccnt 0x0" ::: "memory");
    __syncthreads();

    // ---- cooperative gaussian-basis tile: 64m x 128k, 32 bases/thread ----
    {
      constexpr float kC[8] = {-0.044f, -0.036f, -0.028f, -0.020f,
                               -0.012f, -0.004f,  0.004f,  0.012f};
      float4 xv = *(const float4*)&lds_x[xrow][xq * 4];
      float xs[4] = {xv.x, xv.y, xv.z, xv.w};
#pragma unroll
      for (int j = 0; j < 4; ++j) {
        float x = xs[j];
        union { bf16_t h[8]; uint4 q; } w;
#pragma unroll
        for (int g = 0; g < 8; ++g) {
          float d = x - kC[g];
          w.h[g] = (bf16_t)__builtin_amdgcn_exp2f((d * KEXP2) * d);  // TRANS
        }
        *(uint4*)&lds_bas[xrow][xq * 32 + j * 8] = w.q;
      }
    }
    __syncthreads();

    // ---- L2 prefetch of next chunk's sw tile (overlaps this chunk's WMMAs) ----
    if (i0 + 16 < I) {
#pragma unroll
      for (int pf = 0; pf < 2; ++pf) {
        int l   = tid + pf * 256;      // 512 x 128B lines = next 64KB sw tile
        int ii  = l >> 5;              // feature row
        int off = (l & 31) << 5;       // 32-float (128B) line within row span
        __builtin_prefetch(&SW[(((size_t)(i0 + 16 + ii) * O) + blockN) * 8 + off], 0, 3);
      }
    }

    // ---- base GEMM piece (K=16 features, zero-padded to WMMA K=32) ----
    // A layout (16-bit 16x32): lane<16 elem e -> K = e<8 ? e : e+8 ;
    //                          lane>=16      -> K = e<8 ? e+8 : e+16.
    // Nonzero only where K<16  <=>  e<8; feature = e + 8*lhalf.
    v16bf abase[2];
#pragma unroll
    for (int mt = 0; mt < 2; ++mt) {
      int m = wm + mt * 16 + l16;
      v16bf a;
#pragma unroll
      for (int e = 0; e < 8; ++e) a[e] = (bf16_t)lds_x[m][e + (lhalf << 3)];
#pragma unroll
      for (int e = 8; e < 16; ++e) a[e] = (bf16_t)0.0f;
      abase[mt] = a;
    }
#pragma unroll
    for (int nt = 0; nt < 2; ++nt) {
      int n = wn + nt * 16 + l16;
      v16bf b;  // B layout: elem e<8 -> K = e + 8*lhalf (valid), e>=8 -> K>=16 (zero pad)
#pragma unroll
      for (int e = 0; e < 16; ++e) b[e] = (e < 8) ? lds_bw[n][e + (lhalf << 3)] : (bf16_t)0.0f;
#pragma unroll
      for (int mt = 0; mt < 2; ++mt)
        acc_b[mt][nt] = __builtin_amdgcn_wmma_f32_16x16x32_bf16(
            false, abase[mt], false, b, (short)0, acc_b[mt][nt], false, false);
    }

    // ---- spline GEMM: K-chunk 128 = 4 WMMAs of K=32; A from LDS basis tile ----
#pragma unroll
    for (int kp = 0; kp < 4; ++kp) {
      v16bf asp[2];
#pragma unroll
      for (int mt = 0; mt < 2; ++mt) {
        int m  = wm + mt * 16 + l16;
        int ka = kp * 32 + (lhalf << 3);  // elems 0..7 <- K=ka.., elems 8..15 <- K=ka+16..
        union { uint4 q[2]; v16bf v; } u;
        u.q[0] = *(const uint4*)&lds_bas[m][ka];
        u.q[1] = *(const uint4*)&lds_bas[m][ka + 16];
        asp[mt] = u.v;
      }
#pragma unroll
      for (int nt = 0; nt < 2; ++nt) {
        int n  = wn + nt * 16 + l16;
        int kb = kp * 32 + (lhalf << 3);  // elems 0..7 <- K=kb.., elems 8..15 <- K=kb+16..
        union { uint4 q[2]; v16bf v; } u;
        u.q[0] = *(const uint4*)&lds_sw[n][kb];
        u.q[1] = *(const uint4*)&lds_sw[n][kb + 16];
#pragma unroll
        for (int mt = 0; mt < 2; ++mt)
          acc_s[mt][nt] = __builtin_amdgcn_wmma_f32_16x16x32_bf16(
              false, asp[mt], false, u.v, (short)0, acc_s[mt][nt], false, false);
      }
    }
  }

  // ---- epilogue: silu(base)*sb + spline*ss + noise*sn ----
  const float sn = SN[0];
#pragma unroll
  for (int mt = 0; mt < 2; ++mt)
#pragma unroll
    for (int nt = 0; nt < 2; ++nt) {
      int ncol = blockN + wn + nt * 16 + l16;
      float sb = SB[ncol];
      float ss = SS[ncol];
#pragma unroll
      for (int v = 0; v < 8; ++v) {
        // C/D layout: VGPR v -> M = v + 8*lhalf, N = lane%16
        int mrow = blockM + wm + mt * 16 + v + (lhalf << 3);
        float u    = acc_b[mt][nt][v];
        float base = u / (1.0f + __expf(-u));  // silu
        float nz   = gauss_noise(seed, (unsigned)(mrow * O + ncol));
        OUT[(size_t)mrow * O + ncol] = base * sb + acc_s[mt][nt][v] * ss + nz * sn;
      }
    }
}

extern "C" void kernel_launch(void* const* d_in, const int* in_sizes, int n_in,
                              void* d_out, int out_size, void* d_ws, size_t ws_size,
                              hipStream_t stream) {
  (void)in_sizes; (void)n_in; (void)out_size; (void)ws_size;
  const int B = 8192;
  const float* x = (const float*)d_in[0];
  const float *sw[4], *bw[4], *sb[4], *ss[4], *sn[4];
  for (int l = 0; l < 4; ++l) {
    sw[l] = (const float*)d_in[1 + 5 * l];
    bw[l] = (const float*)d_in[2 + 5 * l];
    sb[l] = (const float*)d_in[3 + 5 * l];
    ss[l] = (const float*)d_in[4 + 5 * l];
    sn[l] = (const float*)d_in[5 + 5 * l];
  }
  float* h    = (float*)d_ws;                        // [8192,2048] scratch (reused)
  float* xrec = (float*)d_out;                       // [8192,1024]
  float* z    = (float*)d_out + (size_t)B * 1024;    // [8192,256] (second output)

  dim3 blk(256);
  auto launch = [&](const float* in, int I, int O, int l, float* out) {
    dim3 grd(B / 64, O / 128);
    kan_layer_kernel<<<grd, blk, 0, stream>>>(in, sw[l], bw[l], sb[l], ss[l], sn[l],
                                              out, I, O, 0x1234567u * (unsigned)(l + 1));
  };
  launch(x, 1024, 2048, 0, h);     // layer 0: x -> h
  launch(h, 2048, 256,  1, z);     // layer 1: h -> z (straight into d_out region 2)
  launch(z, 256,  2048, 2, h);     // layer 2: z -> h (ws reused)
  launch(h, 2048, 1024, 3, xrec);  // layer 3: h -> x_rec
}
